// SKSCAM_601295421851
// MI455X (gfx1250) — matrix operand order repaired and checked by reading, not compiled, not run on hardware
//
#include <hip/hip_runtime.h>
#include <hip/hip_bf16.h>

// ---------------------------------------------------------------------------
// SKSCAM for MI455X (gfx1250): all heavy math on v_wmma_f32_16x16x32_bf16.
// ---------------------------------------------------------------------------

#define B_    4
#define C_    64
#define H_    96
#define W_    256
#define M_    4
#define PAD_  8
#define HP_   (H_ + 2 * PAD_)   // 112
#define WP_   (W_ + 2 * PAD_)   // 272
#define NPIX_ (B_ * H_ * W_)    // 98304

typedef __bf16 bf16;
typedef __attribute__((ext_vector_type(16))) __bf16 bf16x16;
typedef __attribute__((ext_vector_type(8)))  __bf16 bf16x8;
typedef __attribute__((ext_vector_type(8)))  float  f32x8;

__device__ __forceinline__ f32x8 zero8() {
  f32x8 z = {0.f, 0.f, 0.f, 0.f, 0.f, 0.f, 0.f, 0.f};
  return z;
}

__device__ __forceinline__ f32x8 wmma_bf16(bf16x16 a, bf16x16 b, f32x8 c) {
  // D(16x16,f32) = A(16x32,bf16) x B(32x16,bf16) + C
  return __builtin_amdgcn_wmma_f32_16x16x32_bf16(false, a, false, b,
                                                 (short)0, c, false, false);
}

// A-fragment (16x32 bf16) element -> K index for element e (0..15), lane l:
//   K = 16*(l>=16) + e (contiguous, same as B), M = l&15
// B-fragment (32x16 bf16): lane l reads K = 16*(l>=16) + e (contiguous), N = l&15
// C/D  (16x16 f32): VGPR r holds M = r + 8*(l>=16), N = l&15

// ---------------------------------------------------------------------------
// 1) LayerNorm2d -> padded channels-last bf16 (Xn) + raw channels-last (Xcl)
// ---------------------------------------------------------------------------
__global__ __launch_bounds__(256) void ln_prep_kernel(
    const float* __restrict__ x, const float* __restrict__ g,
    const float* __restrict__ be, bf16* __restrict__ Xn,
    bf16* __restrict__ Xcl) {
  int idx = blockIdx.x * blockDim.x + threadIdx.x;
  const int total = B_ * HP_ * WP_;
  if (idx >= total) return;
  int wp = idx % WP_;
  int hp = (idx / WP_) % HP_;
  int b  = idx / (WP_ * HP_);
  bf16* outp = Xn + (size_t)idx * C_;
  int h = hp - PAD_, w = wp - PAD_;
  if (h < 0 || h >= H_ || w < 0 || w >= W_) {
    bf16x8 z8;
#pragma unroll
    for (int r = 0; r < 8; r++) z8[r] = (bf16)0.f;
#pragma unroll
    for (int cg = 0; cg < 8; cg++) *(bf16x8*)(outp + cg * 8) = z8;
    return;
  }
  float v[C_];
  float s = 0.f, ss = 0.f;
#pragma unroll
  for (int c = 0; c < C_; c++) {
    float t = x[(((size_t)b * C_ + c) * H_ + h) * W_ + w];
    v[c] = t;
    s += t;
    ss += t * t;
  }
  float mu  = s * (1.f / C_);
  float var = ss * (1.f / C_) - mu * mu;
  float inv = rsqrtf(var + 1e-6f);
  bf16* rawp = Xcl + ((size_t)(b * H_ + h) * W_ + w) * C_;
#pragma unroll
  for (int cg = 0; cg < 8; cg++) {
    bf16x8 tn, tr;
#pragma unroll
    for (int r = 0; r < 8; r++) {
      int c  = cg * 8 + r;
      tn[r] = (bf16)((v[c] - mu) * inv * g[c] + be[c]);
      tr[r] = (bf16)v[c];
    }
    *(bf16x8*)(outp + cg * 8) = tn;
    *(bf16x8*)(rawp + cg * 8) = tr;
  }
}

// ---------------------------------------------------------------------------
// 2) Dilated 3x3 conv branches (implicit GEMM on WMMA).
//    Block: 256 thr = 8 waves covering 2 rows x full W.
//    Wave: 64 pixels (4 tiles) x 32 couts; tap loop FULLY UNROLLED so the 8
//    accumulator tuples stay in fixed even-aligned VGPRs (no copies/NOPs).
//    grid = (1, H/2, B*M*2).  Accumulates global-pool Z via shuffles+atomics,
//    stores branch output D (bf16, channels-last).
// ---------------------------------------------------------------------------
__global__ __launch_bounds__(256) void skm_conv_kernel(
    const bf16* __restrict__ Xn, const float* __restrict__ dw,
    const float* __restrict__ db, bf16* __restrict__ D,
    float* __restrict__ Z) {
  __shared__ bf16 wpack[36 * 512];  // 36 A-fragments * 32 lanes * 16 elems
  __shared__ float biasS[C_];
  int z  = blockIdx.z;
  int ch = z & 1;           // cout half (0..1)
  int m  = (z >> 1) & 3;    // branch
  int b  = z >> 3;          // batch
  int tid = threadIdx.x;

  // pack weights into A-fragment layout (bf16)
  const float* wsrc = dw + (size_t)m * C_ * C_ * 9;
  for (int i = tid; i < 36 * 512; i += 256) {
    int e    = i & 15;
    int l    = (i >> 4) & 31;
    int frag = i >> 9;          // (t*2+kk)*2+cb
    int cb   = frag & 1;
    int kk   = (frag >> 1) & 1;
    int t    = frag >> 2;       // 0..8
    int vv   = e >> 1, lo = e & 1;
    int cout = 32 * ch + 16 * cb + (l & 15);
    int cin  = 32 * kk + ((l >= 16) ? 16 : 0) + e;
    int ky = t / 3, kx = t % 3;
    wpack[i] = (bf16)wsrc[((cout * C_ + cin) * 3 + ky) * 3 + kx];
  }
  if (tid < C_) biasS[tid] = db[m * C_ + tid];
  __syncthreads();

  int wave = tid >> 5, lane = tid & 31;
  int p = lane & 15, hi = lane >> 4;
  int h  = blockIdx.y * 2 + (wave >> 2);  // 2 rows per block
  int w0 = (wave & 3) * 64;               // 4 tiles of 16 pixels per wave
  int d  = 2 * (m + 1);

  f32x8 acc[2][4];
#pragma unroll
  for (int cb = 0; cb < 2; cb++)
#pragma unroll
    for (int tl = 0; tl < 4; tl++) acc[cb][tl] = zero8();

#pragma unroll
  for (int t = 0; t < 9; t++) {
    int dy = (t / 3 - 1) * d;
    int dx = (t % 3 - 1) * d;
    int hh = h + PAD_ + dy;
    const bf16* bp =
        Xn + (((size_t)b * HP_ + hh) * WP_ + (w0 + p + PAD_ + dx)) * C_ +
        16 * hi;
#pragma unroll
    for (int kk = 0; kk < 2; kk++) {
      bf16x16 a0 =
          *(const bf16x16*)(&wpack[(((t * 2 + kk) * 2 + 0) * 32 + lane) * 16]);
      bf16x16 a1 =
          *(const bf16x16*)(&wpack[(((t * 2 + kk) * 2 + 1) * 32 + lane) * 16]);
#pragma unroll
      for (int tl = 0; tl < 4; tl++) {
        // tile stride = 16 pixels * 64 ch = 1024 bf16 (constant ioffset)
        bf16x16 bfrag = *(const bf16x16*)(bp + kk * 32 + tl * 1024);
        acc[0][tl] = wmma_bf16(a0, bfrag, acc[0][tl]);
        acc[1][tl] = wmma_bf16(a1, bfrag, acc[1][tl]);
      }
    }
  }

  // bias add + store D (bf16 channels-last) + pooled-sum accumulation
  bf16* dout = D + ((((size_t)m * B_ + b) * H_ + h) * W_ + w0) * C_;
  float zsum[2][8];
#pragma unroll
  for (int cb = 0; cb < 2; cb++)
#pragma unroll
    for (int r = 0; r < 8; r++) zsum[cb][r] = 0.f;

#pragma unroll
  for (int tl = 0; tl < 4; tl++) {
#pragma unroll
    for (int cb = 0; cb < 2; cb++) {
      bf16x8 pack;
#pragma unroll
      for (int r = 0; r < 8; r++) {
        float val = acc[cb][tl][r] + biasS[32 * ch + 16 * cb + 8 * hi + r];
        zsum[cb][r] += val;
        pack[r] = (bf16)val;
      }
      *(bf16x8*)(dout + (tl * 16 + p) * C_ + 32 * ch + 16 * cb + 8 * hi) = pack;
    }
  }
#pragma unroll
  for (int cb = 0; cb < 2; cb++) {
#pragma unroll
    for (int r = 0; r < 8; r++) {
      float v = zsum[cb][r];
      v += __shfl_xor(v, 1);
      v += __shfl_xor(v, 2);
      v += __shfl_xor(v, 4);
      v += __shfl_xor(v, 8);
      if ((lane & 15) == 0)
        atomicAdd(&Z[b * C_ + 32 * ch + 16 * cb + 8 * hi + r], v);
    }
  }
}

// ---------------------------------------------------------------------------
// 3) SKM gating: Z -> relu FC -> FC -> softmax over channels per branch
// ---------------------------------------------------------------------------
__global__ __launch_bounds__(128) void skm_weights_kernel(
    const float* __restrict__ Z, const float* __restrict__ c1w,
    const float* __restrict__ c1b, const float* __restrict__ pw,
    float* __restrict__ Sw) {
  __shared__ float Zs[C_], Ss[2 * C_], Ps[M_ * C_], red[8];
  int b = blockIdx.x, tid = threadIdx.x;
  if (tid < C_) Zs[tid] = Z[b * C_ + tid] * (1.0f / (H_ * W_));
  __syncthreads();
  float s = c1b[tid];
  for (int c = 0; c < C_; c++) s += Zs[c] * c1w[tid * C_ + c];
  Ss[tid] = fmaxf(s, 0.f);
  __syncthreads();
  for (int k = tid; k < M_ * C_; k += 128) {
    float p = 0.f;
    for (int j = 0; j < 2 * C_; j++) p += Ss[j] * pw[j * (M_ * C_) + k];
    Ps[k] = p;
  }
  __syncthreads();
  if (tid < M_) {
    float mx = -3.4e38f;
    for (int c = 0; c < C_; c++) mx = fmaxf(mx, Ps[tid * C_ + c]);
    float sm = 0.f;
    for (int c = 0; c < C_; c++) sm += __expf(Ps[tid * C_ + c] - mx);
    red[tid] = mx;
    red[4 + tid] = 1.f / sm;
  }
  __syncthreads();
  for (int k = tid; k < M_ * C_; k += 128) {
    int m = k >> 6;
    Sw[(b * M_ + m) * C_ + (k & 63)] = __expf(Ps[k] - red[m]) * red[4 + m];
  }
}

// ---------------------------------------------------------------------------
// 4) Q = sum_m D[m] * Sw[b,m,c]   (elementwise, channels-last bf16 out)
// ---------------------------------------------------------------------------
__global__ __launch_bounds__(256) void skm_combine_kernel(
    const bf16* __restrict__ D, const float* __restrict__ Sw,
    bf16* __restrict__ Q) {
  size_t idx = (size_t)blockIdx.x * blockDim.x + threadIdx.x;
  if (idx >= (size_t)NPIX_ * 8) return;
  int cg = (int)(idx & 7);
  size_t pix = idx >> 3;
  int b = (int)(pix / (H_ * W_));
  float accv[8] = {0.f, 0.f, 0.f, 0.f, 0.f, 0.f, 0.f, 0.f};
#pragma unroll
  for (int m = 0; m < M_; m++) {
    bf16x8 dv = *(const bf16x8*)(D + ((size_t)m * NPIX_ + pix) * C_ + cg * 8);
    const float* sw = Sw + (b * M_ + m) * C_ + cg * 8;
#pragma unroll
    for (int r = 0; r < 8; r++) accv[r] += (float)dv[r] * sw[r];
  }
  bf16x8 o;
#pragma unroll
  for (int r = 0; r < 8; r++) o[r] = (bf16)accv[r];
  *(bf16x8*)(Q + pix * C_ + cg * 8) = o;
}

// ---------------------------------------------------------------------------
// 5) V = x @ w2^T + b2 (1x1 conv) on WMMA, stored transposed Vt[b,h,c,w]
// ---------------------------------------------------------------------------
__global__ __launch_bounds__(256) void vproj_kernel(
    const bf16* __restrict__ Xcl, const float* __restrict__ w2,
    const float* __restrict__ b2, bf16* __restrict__ Vt) {
  __shared__ bf16 wpack[8 * 512];
  __shared__ float biasS[C_];
  int tid = threadIdx.x;
  for (int i = tid; i < 8 * 512; i += 256) {
    int e = i & 15, l = (i >> 4) & 31, frag = i >> 9;
    int cb = frag & 3, kk = frag >> 2;
    int vv = e >> 1, lo = e & 1;
    int cout = 16 * cb + (l & 15);
    int cin  = 32 * kk + ((l >= 16) ? 16 : 0) + e;
    wpack[i] = (bf16)w2[cout * C_ + cin];
  }
  if (tid < C_) biasS[tid] = b2[tid];
  __syncthreads();

  int wave = tid >> 5, lane = tid & 31;
  int p = lane & 15, hi = lane >> 4;
  int b = blockIdx.z, h = blockIdx.y;
  int w0 = (blockIdx.x * 8 + wave) * 16;
  f32x8 acc[4];
#pragma unroll
  for (int cb = 0; cb < 4; cb++) acc[cb] = zero8();
  const bf16* bp = Xcl + (((size_t)b * H_ + h) * W_ + w0 + p) * C_ + 16 * hi;
#pragma unroll
  for (int kk = 0; kk < 2; kk++) {
    bf16x16 bfrag = *(const bf16x16*)(bp + 32 * kk);
#pragma unroll
    for (int cb = 0; cb < 4; cb++) {
      bf16x16 afrag = *(const bf16x16*)(&wpack[((kk * 4 + cb) * 32 + lane) * 16]);
      acc[cb] = wmma_bf16(afrag, bfrag, acc[cb]);
    }
  }
  bf16* vt = Vt + ((size_t)b * H_ + h) * C_ * W_;
#pragma unroll
  for (int cb = 0; cb < 4; cb++) {
#pragma unroll
    for (int r = 0; r < 8; r++) {
      int c = 16 * cb + 8 * hi + r;
      vt[(size_t)c * W_ + w0 + p] = (bf16)(acc[cb][r] + biasS[c]);
    }
  }
}

// ---------------------------------------------------------------------------
// 6) Row attention: one wave per (b,h,16-query tile).
//    S = Q K^T * scale (WMMA, K-dim = channels); full-row softmax in LDS
//    (unnormalized; 1/sum folded into epilogue); O = P V (WMMA, K-dim = keys);
//    out = x + O * blend  (channels-first f32).
// ---------------------------------------------------------------------------
__global__ __launch_bounds__(32) void attn_kernel(
    const bf16* __restrict__ Q, const bf16* __restrict__ K,
    const bf16* __restrict__ Vt, const float* __restrict__ x,
    const float* __restrict__ blend, float* __restrict__ out) {
  __shared__ float S[16 * W_];
  __shared__ float invsum[16];
  int lane = threadIdx.x;
  int qt = blockIdx.x, h = blockIdx.y, b = blockIdx.z;
  int p = lane & 15, hi = lane >> 4;
  const float scale = 0.125f;  // C^-0.5, C=64

  // Q A-fragments (this wave's 16 queries, 2 cin chunks), kept in registers
  const bf16* qrow = Q + ((size_t)b * H_ + h) * W_ * C_;
  bf16x16 aq[2];
#pragma unroll
  for (int kk = 0; kk < 2; kk++) {
#pragma unroll
    for (int v = 0; v < 8; v++) {
      int c = 32 * kk + 16 * hi + 2 * v;
      const bf16* qp = qrow + (size_t)(qt * 16 + p) * C_ + c;
      aq[kk][2 * v]     = qp[0];
      aq[kk][2 * v + 1] = qp[1];
    }
  }

  // scores
  const bf16* krow = K + ((size_t)b * H_ + h) * W_ * C_;
  for (int jt = 0; jt < 16; jt++) {
    f32x8 acc = zero8();
    const bf16* kp = krow + (size_t)(jt * 16 + p) * C_ + 16 * hi;
#pragma unroll
    for (int kk = 0; kk < 2; kk++) {
      bf16x16 bfrag = *(const bf16x16*)(kp + 32 * kk);
      acc = wmma_bf16(aq[kk], bfrag, acc);
    }
#pragma unroll
    for (int r = 0; r < 8; r++)
      S[(r + 8 * hi) * W_ + jt * 16 + p] = acc[r] * scale;
  }
  __syncthreads();

  // row softmax (rows = p, halves split columns; exp stored unnormalized)
  {
    int row = p;
    float mx = -3.4e38f;
    for (int j = hi * 128; j < hi * 128 + 128; j++)
      mx = fmaxf(mx, S[row * W_ + j]);
    mx = fmaxf(mx, __shfl_xor(mx, 16));
    float sm = 0.f;
    for (int j = hi * 128; j < hi * 128 + 128; j++) {
      float e = __expf(S[row * W_ + j] - mx);
      S[row * W_ + j] = e;
      sm += e;
    }
    sm += __shfl_xor(sm, 16);
    if (hi == 0) invsum[row] = 1.f / sm;
  }
  __syncthreads();

  // O = P @ V
  f32x8 oacc[4];
#pragma unroll
  for (int ct = 0; ct < 4; ct++) oacc[ct] = zero8();
  const bf16* vrow = Vt + ((size_t)b * H_ + h) * C_ * W_;
  for (int jk = 0; jk < 8; jk++) {
    bf16x16 ap;
#pragma unroll
    for (int v = 0; v < 8; v++) {
      int j = 32 * jk + 16 * hi + 2 * v;
      ap[2 * v]     = (bf16)S[p * W_ + j];
      ap[2 * v + 1] = (bf16)S[p * W_ + j + 1];
    }
#pragma unroll
    for (int ct = 0; ct < 4; ct++) {
      int c = 16 * ct + p;
      bf16x16 bfrag = *(const bf16x16*)(vrow + (size_t)c * W_ + 32 * jk + 16 * hi);
      oacc[ct] = wmma_bf16(ap, bfrag, oacc[ct]);
    }
  }

  // epilogue: out[b,c,h,wq] = x + O * invsum[row] * blend[c]
#pragma unroll
  for (int ct = 0; ct < 4; ct++) {
    int c = 16 * ct + p;
    float bl = blend[c];
#pragma unroll
    for (int r = 0; r < 8; r++) {
      int i  = r + 8 * hi;
      int wq = qt * 16 + i;
      size_t off = (((size_t)b * C_ + c) * H_ + h) * W_ + wq;
      out[off] = x[off] + oacc[ct][r] * invsum[i] * bl;
    }
  }
}

// ---------------------------------------------------------------------------
__global__ void zero_kernel(float* p, int n) {
  int i = blockIdx.x * blockDim.x + threadIdx.x;
  if (i < n) p[i] = 0.f;
}

// ---------------------------------------------------------------------------
extern "C" void kernel_launch(void* const* d_in, const int* in_sizes, int n_in,
                              void* d_out, int out_size, void* d_ws,
                              size_t ws_size, hipStream_t stream) {
  const float* x_l   = (const float*)d_in[0];
  const float* x_r   = (const float*)d_in[1];
  const float* lnl_g = (const float*)d_in[2];
  const float* lnl_b = (const float*)d_in[3];
  const float* lnr_g = (const float*)d_in[4];
  const float* lnr_b = (const float*)d_in[5];
  const float* l_dw  = (const float*)d_in[6];
  const float* l_db  = (const float*)d_in[7];
  const float* l_c1w = (const float*)d_in[8];
  const float* l_c1b = (const float*)d_in[9];
  const float* l_pw  = (const float*)d_in[10];
  const float* r_dw  = (const float*)d_in[11];
  const float* r_db  = (const float*)d_in[12];
  const float* r_c1w = (const float*)d_in[13];
  const float* r_c1b = (const float*)d_in[14];
  const float* r_pw  = (const float*)d_in[15];
  const float* l2w   = (const float*)d_in[16];
  const float* l2b   = (const float*)d_in[17];
  const float* r2w   = (const float*)d_in[18];
  const float* r2b   = (const float*)d_in[19];
  const float* beta  = (const float*)d_in[20];
  const float* gamma = (const float*)d_in[21];
  float* out = (float*)d_out;

  // workspace carve-out (~208 MB total)
  char* ws = (char*)d_ws;
  size_t off = 0;
  auto alloc = [&](size_t bytes) {
    char* p = ws + off;
    off = (off + bytes + 255) & ~(size_t)255;
    return p;
  };
  float* Zb    = (float*)alloc(2 * B_ * C_ * sizeof(float));
  float* Swb   = (float*)alloc(2 * B_ * M_ * C_ * sizeof(float));
  bf16* Xn[2]  = {(bf16*)alloc((size_t)B_ * HP_ * WP_ * C_ * 2),
                  (bf16*)alloc((size_t)B_ * HP_ * WP_ * C_ * 2)};
  bf16* Xcl[2] = {(bf16*)alloc((size_t)NPIX_ * C_ * 2),
                  (bf16*)alloc((size_t)NPIX_ * C_ * 2)};
  bf16* Db[2]  = {(bf16*)alloc((size_t)M_ * NPIX_ * C_ * 2),
                  (bf16*)alloc((size_t)M_ * NPIX_ * C_ * 2)};
  bf16* Qb[2]  = {(bf16*)alloc((size_t)NPIX_ * C_ * 2),
                  (bf16*)alloc((size_t)NPIX_ * C_ * 2)};
  bf16* Vt[2]  = {(bf16*)alloc((size_t)NPIX_ * C_ * 2),
                  (bf16*)alloc((size_t)NPIX_ * C_ * 2)};
  (void)ws_size; (void)in_sizes; (void)n_in; (void)out_size;

  // 0) zero global-pool accumulators (fresh every call: graph-replay safe)
  zero_kernel<<<2, 256, 0, stream>>>(Zb, 2 * B_ * C_);

  const int lnBlocks = (B_ * HP_ * WP_ + 255) / 256;
  // 1) LN + channels-last staging
  ln_prep_kernel<<<lnBlocks, 256, 0, stream>>>(x_l, lnl_g, lnl_b, Xn[0], Xcl[0]);
  ln_prep_kernel<<<lnBlocks, 256, 0, stream>>>(x_r, lnr_g, lnr_b, Xn[1], Xcl[1]);

  // 2) dilated conv branches (WMMA): block covers 2 rows x full W
  dim3 cgrid(1, H_ / 2, B_ * M_ * 2);
  skm_conv_kernel<<<cgrid, 256, 0, stream>>>(Xn[0], l_dw, l_db, Db[0], Zb);
  skm_conv_kernel<<<cgrid, 256, 0, stream>>>(Xn[1], r_dw, r_db, Db[1],
                                             Zb + B_ * C_);

  // 3) gating weights
  skm_weights_kernel<<<B_, 128, 0, stream>>>(Zb, l_c1w, l_c1b, l_pw, Swb);
  skm_weights_kernel<<<B_, 128, 0, stream>>>(Zb + B_ * C_, r_c1w, r_c1b, r_pw,
                                             Swb + B_ * M_ * C_);

  // 4) branch combine -> Q (bf16, channels-last)
  const int combBlocks = (NPIX_ * 8 + 255) / 256;
  skm_combine_kernel<<<combBlocks, 256, 0, stream>>>(Db[0], Swb, Qb[0]);
  skm_combine_kernel<<<combBlocks, 256, 0, stream>>>(Db[1], Swb + B_ * M_ * C_,
                                                     Qb[1]);

  // 5) V projections (WMMA), transposed store for attention B-fragments
  dim3 vgrid(W_ / 128, H_, B_);
  vproj_kernel<<<vgrid, 256, 0, stream>>>(Xcl[0], l2w, l2b, Vt[0]);
  vproj_kernel<<<vgrid, 256, 0, stream>>>(Xcl[1], r2w, r2b, Vt[1]);

  // 6) cross attention, both directions (axis=-2 softmax == swapped roles)
  dim3 agrid(W_ / 16, H_, B_);
  attn_kernel<<<agrid, 32, 0, stream>>>(Qb[0], Qb[1], Vt[1], x_l, beta, out);
  attn_kernel<<<agrid, 32, 0, stream>>>(Qb[1], Qb[0], Vt[0], x_r, gamma,
                                        out + (size_t)B_ * C_ * H_ * W_);
}
